// QuantLinearW4A4Tl_68788196212774
// MI455X (gfx1250) — 8-bit matrix operand order repaired and checked by reading, compiled, not run on hardware
//
#include <hip/hip_runtime.h>

typedef int   v8i __attribute__((ext_vector_type(8)));
typedef float v8f __attribute__((ext_vector_type(8)));
typedef float v2f __attribute__((ext_vector_type(2)));

namespace {
constexpr int MDIM = 8192, KDIM = 4096, NDIM = 4096, RDIM = 32;
constexpr int KP   = KDIM / 2;
constexpr int BM = 128, BN = 128, BK = 64;
constexpr int STAGES = KDIM / BK;
constexpr int LDA    = BK + 16;
constexpr int SM_AB  = BM * LDA;
constexpr int SMEM_BYTES = 2 * SM_AB + 5 * 128 * 4;
}

__device__ __forceinline__ void unpack8(const int4 p, unsigned &d0, unsigned &d1) {
  unsigned t0 = __builtin_amdgcn_perm((unsigned)p.y, (unsigned)p.x, 0x00000400u);
  unsigned t1 = __builtin_amdgcn_perm((unsigned)p.w, (unsigned)p.z, 0x04000000u);
  unsigned pk = __builtin_amdgcn_perm(t1, t0, 0x07060100u);
  pk ^= 0x88888888u;
  unsigned lo =  pk       & 0x0F0F0F0Fu;
  unsigned hi = (pk >> 4) & 0x0F0F0F0Fu;
  d0 = __builtin_amdgcn_perm(hi, lo, 0x05010400u);
  d1 = __builtin_amdgcn_perm(hi, lo, 0x07030602u);
}

__global__ __launch_bounds__(256) void rowsum_kernel(const int* __restrict__ pk,
                                                     int* __restrict__ out, int dpr) {
  __shared__ int red[256];
  const int row = blockIdx.x, tid = threadIdx.x;
  const int* p = pk + (size_t)row * dpr;
  int s = 0;
  for (int j = tid; j < dpr; j += 256) {
    unsigned b = ((unsigned)p[j] & 0xFFu) ^ 0x88u;
    s += (int)((b & 0xFu) + ((b >> 4) & 0xFu));
  }
  red[tid] = s; __syncthreads();
  for (int off = 128; off > 0; off >>= 1) {
    if (tid < off) red[tid] += red[tid + off];
    __syncthreads();
  }
  if (tid == 0) out[row] = red[0];
}

__global__ __launch_bounds__(256) void w4a4_gemm_kernel(
    const int* __restrict__ xq,  const float* __restrict__ xs,
    const int* __restrict__ wq,  const float* __restrict__ ws,
    const float* __restrict__ bias, const float* __restrict__ xr,
    const float* __restrict__ wo,   const int* __restrict__ rsx,
    const int* __restrict__ rsw,    float* __restrict__ out)
{
  extern __shared__ char smem[];
  char*  As   = smem;
  char*  Bs   = smem + SM_AB;
  float* sx_s = (float*)(smem + 2 * SM_AB);
  float* sw_s = sx_s + BM;
  float* bi_s = sw_s + BN;
  int*   rx_s = (int*)(bi_s + BN);
  int*   rw_s = rx_s + BM;

  const int tid = threadIdx.x;
  const int BM0 = blockIdx.y * BM;
  const int BN0 = blockIdx.x * BN;

  if (tid < 128) {
    sx_s[tid] = xs[BM0 + tid];
    rx_s[tid] = rsx[BM0 + tid];
    sw_s[tid] = ws[BN0 + tid];
    bi_s[tid] = bias[BN0 + tid];
    rw_s[tid] = rsw[BN0 + tid];
  }

  const int srow = tid >> 3;
  const int pcol = (tid & 7) * 4;
  const int soff = (tid & 7) * 8;

  int4 pa[4], pb[4];
  auto loadStage = [&](int s) {
#pragma unroll
    for (int i = 0; i < 4; ++i) {
      pa[i] = *(const int4*)(xq + (size_t)(BM0 + i * 32 + srow) * KP + s * (BK / 2) + pcol);
      pb[i] = *(const int4*)(wq + (size_t)(BN0 + i * 32 + srow) * KP + s * (BK / 2) + pcol);
    }
  };

  const int lane = tid & 31;
  const int l    = lane & 15;
  const int hl   = lane >> 4;
  const int wv   = tid >> 5;
  const int wm   = wv >> 1;
  const int wn   = wv & 1;

  v8i acc[2][4] = {};

  loadStage(0);
#pragma unroll 1
  for (int s = 0; s < STAGES; ++s) {
    __syncthreads();
#pragma unroll
    for (int i = 0; i < 4; ++i) {
      unsigned d0, d1;
      unpack8(pa[i], d0, d1);
      *(int2*)(As + (i * 32 + srow) * LDA + soff) = make_int2((int)d0, (int)d1);
      unpack8(pb[i], d0, d1);
      *(int2*)(Bs + (i * 32 + srow) * LDA + soff) = make_int2((int)d0, (int)d1);
    }
    __syncthreads();
    if (s + 1 < STAGES) loadStage(s + 1);
    if (s + 2 < STAGES) {
      __builtin_prefetch(xq + (size_t)(BM0 + srow) * KP + (s + 2) * (BK / 2) + pcol, 0, 3);
      __builtin_prefetch(wq + (size_t)(BN0 + srow) * KP + (s + 2) * (BK / 2) + pcol, 0, 3);
    }

    v8i afr[2];
#pragma unroll
    for (int ti = 0; ti < 2; ++ti) {
      const char* rp = As + (wm * 32 + ti * 16 + l) * LDA;
#pragma unroll
      for (int p = 0; p < 4; ++p) {
        int2 d = *(const int2*)(rp + (p >> 1) * 32 + hl * 16 + (p & 1) * 8);
        afr[ti][2 * p]     = d.x;
        afr[ti][2 * p + 1] = d.y;
      }
    }
#pragma unroll
    for (int tj = 0; tj < 4; ++tj) {
      const char* rp = Bs + (wn * 64 + tj * 16 + l) * LDA;
      int4 u0 = *(const int4*)(rp + hl * 16);
      int4 u1 = *(const int4*)(rp + 32 + hl * 16);
      v8i bfr;
      bfr[0] = u0.x; bfr[1] = u0.y; bfr[2] = u0.z; bfr[3] = u0.w;
      bfr[4] = u1.x; bfr[5] = u1.y; bfr[6] = u1.z; bfr[7] = u1.w;
#pragma unroll
      for (int ti = 0; ti < 2; ++ti)
        acc[ti][tj] = __builtin_amdgcn_wmma_i32_16x16x64_iu8(
            false, afr[ti], false, bfr, acc[ti][tj], false, false);
    }
  }

  __syncthreads();

#pragma unroll
  for (int ti = 0; ti < 2; ++ti) {
    const int mBase = wm * 32 + ti * 16;
    v2f af[RDIM / 4];
#pragma unroll
    for (int ks = 0; ks < RDIM / 4; ++ks) {
      const float* xp = xr + (size_t)(BM0 + mBase + l) * RDIM + ks * 4 + hl * 2;
      af[ks][0] = xp[0]; af[ks][1] = xp[1];
    }
#pragma unroll
    for (int tj = 0; tj < 4; ++tj) {
      const int   nl  = wn * 64 + tj * 16 + l;
      const float swv = sw_s[nl];
      const float bv  = bi_s[nl];
      const int   rwv = rw_s[nl];
      v8f cf;
#pragma unroll
      for (int r = 0; r < 8; ++r) {
        const int ml   = mBase + r + 8 * hl;
        const int corr = acc[ti][tj][r] + 64 * KDIM - 8 * (rx_s[ml] + rwv);
        cf[r] = (float)corr * sx_s[ml] * swv + bv;
      }
#pragma unroll
      for (int ks = 0; ks < RDIM / 4; ++ks) {
        const float* wp = wo + (size_t)(BN0 + nl) * RDIM + ks * 4 + hl * 2;
        v2f bf; bf[0] = wp[0]; bf[1] = wp[1];
        cf = __builtin_amdgcn_wmma_f32_16x16x4_f32(false, af[ks], false, bf,
                                                   (short)0, cf, false, false);
      }
#pragma unroll
      for (int r = 0; r < 8; ++r) {
        const int gm = BM0 + mBase + r + 8 * hl;
        out[(size_t)gm * NDIM + (BN0 + nl)] = cf[r];
      }
    }
  }
}

extern "C" void kernel_launch(void* const* d_in, const int* in_sizes, int n_in,
                              void* d_out, int out_size, void* d_ws, size_t ws_size,
                              hipStream_t stream) {
  const int*   x_quant   = (const int*)d_in[0];
  const float* x_scale   = (const float*)d_in[1];
  const int*   weight    = (const int*)d_in[2];
  const float* w_scales  = (const float*)d_in[3];
  const float* bias      = (const float*)d_in[4];
  const float* x_r       = (const float*)d_in[5];
  const float* w_outlier = (const float*)d_in[6];
  float*       out       = (float*)d_out;

  int* rsx = (int*)d_ws;
  int* rsw = rsx + MDIM;

  rowsum_kernel<<<MDIM, 256, 0, stream>>>(x_quant, rsx, KP);
  rowsum_kernel<<<NDIM, 256, 0, stream>>>(weight,  rsw, KP);

  dim3 grid(NDIM / BN, MDIM / BM);
  w4a4_gemm_kernel<<<grid, 256, SMEM_BYTES, stream>>>(
      x_quant, x_scale, weight, w_scales, bias, x_r, w_outlier, rsx, rsw, out);
}
